// NT_Xent_Loss_12463995093382
// MI455X (gfx1250) — matrix operand order repaired and checked by reading, not compiled, not run on hardware
//
#include <hip/hip_runtime.h>
#include <hip/hip_bf16.h>
#include <math.h>

typedef __attribute__((ext_vector_type(16))) _Float16 v16h;
typedef __attribute__((ext_vector_type(8)))  _Float16 v8h;
typedef __attribute__((ext_vector_type(4)))  _Float16 v4h;
typedef __attribute__((ext_vector_type(8)))  float    v8f;

#define NHALF   4096          // rows per view
#define NROWS   8192          // 2*NHALF
#define DDIM    512           // embedding dim
#define INV_T   2.0f          // 1 / temperature (T = 0.5)
#define BPAD    (DDIM + 8)    // padded LDS row stride (halfs)
#define TILE_N  32            // columns staged per LDS buffer (2 WMMA sub-tiles)
#define COLSPLIT 2            // column halves (gridDim.y)
#define COLS_PB (NROWS / COLSPLIT)
#define NSTAGES (COLS_PB / TILE_N)
#define EMB_BYTES ((size_t)NROWS * DDIM * 2)

// ---------------------------------------------------------------------------
// CDNA5 async global->LDS copy (ASYNCcnt path), 16 bytes per lane.
// lds_byte_off = low 32 bits of generic shared pointer.
// ---------------------------------------------------------------------------
__device__ __forceinline__ void async_load16(uint32_t lds_byte_off, uint64_t gaddr) {
    asm volatile("global_load_async_to_lds_b128 %0, %1, off"
                 :: "v"(lds_byte_off), "v"(gaddr)
                 : "memory");
}

__device__ __forceinline__ void wait_async0() {
#if __has_builtin(__builtin_amdgcn_s_wait_asynccnt)
    __builtin_amdgcn_s_wait_asynccnt(0);
#else
    asm volatile("s_wait_asynccnt 0x0" ::: "memory");
#endif
}

// ---------------------------------------------------------------------------
// Kernel 1: L2-normalize each row of [z1; z2] and convert to f16 into d_ws.
// ---------------------------------------------------------------------------
__global__ void __launch_bounds__(128)
l2norm_to_f16(const float* __restrict__ z1, const float* __restrict__ z2,
              _Float16* __restrict__ emb) {
    __shared__ float red[4];
    const int row = blockIdx.x;
    const int tid = threadIdx.x;
    const float* src = (row < NHALF) ? (z1 + (size_t)row * DDIM)
                                     : (z2 + (size_t)(row - NHALF) * DDIM);
    float4 v = ((const float4*)src)[tid];
    float ss = v.x * v.x + v.y * v.y + v.z * v.z + v.w * v.w;
#pragma unroll
    for (int off = 16; off > 0; off >>= 1)
        ss += __shfl_xor(ss, off, 32);
    if ((tid & 31) == 0) red[tid >> 5] = ss;
    __syncthreads();
    const float tot   = red[0] + red[1] + red[2] + red[3];
    const float scale = 1.0f / fmaxf(sqrtf(tot), 1e-12f);
    v4h h;
    h[0] = (_Float16)(v.x * scale);
    h[1] = (_Float16)(v.y * scale);
    h[2] = (_Float16)(v.z * scale);
    h[3] = (_Float16)(v.w * scale);
    *(v4h*)(emb + (size_t)row * DDIM + tid * 4) = h;
}

// ---------------------------------------------------------------------------
// Kernel 2: zero the scalar accumulator (d_out is poisoned by the harness).
// ---------------------------------------------------------------------------
__global__ void zero_scalar(float* out) { out[0] = 0.0f; }

// ---------------------------------------------------------------------------
// Kernel 3: fused sim = emb@embT (f16 WMMA, f32 acc) + masked online softmax.
// Grid (64, 2) x 128 threads (4 waves, 1 per SIMD).  Block owns 128 rows
// (32 per wave: TWO resident A strips -> each LDS B-fragment feeds TWO
// accumulating WMMAs, halving DS bytes per FLOP) and one 4096-column half.
// Column tiles are double-buffered in LDS via GLOBAL_LOAD_ASYNC_TO_LDS_B128.
// Per-row partial (m, s, pos) written to workspace; merged by kernel 4.
// ---------------------------------------------------------------------------
__global__ void __launch_bounds__(128)
ntxent_fused(const _Float16* __restrict__ emb,
             float* __restrict__ pm, float* __restrict__ ps,
             float* __restrict__ pp) {
    __shared__ _Float16 btile[2][TILE_N * BPAD];   // 2 x 32.5 KB

    const int tid   = threadIdx.x;
    const int wave  = tid >> 5;        // 0..3
    const int lane  = tid & 31;
    const int l15   = lane & 15;
    const int hi    = lane >> 4;       // half-wave select
    const int mblk  = blockIdx.x * 128;
    const int chalf = blockIdx.y;                  // column half
    const int cbase = chalf * COLS_PB;
    const int arow  = mblk + wave * 32 + l15;      // strip L; strip H = +16

    // --- TWO resident A strips: 2 x 16 K-tiles x v16h = 256 VGPRs.
    // A 16x32 f16 operand, k order used here (same as B): lanes 0-15 hold K [k0..+15];
    // lanes 16-31 hold K [k0+16..+31] (contiguous in memory).
    v16h afragL[16], afragH[16];
    {
        const _Float16* aL = emb + (size_t)arow * DDIM;
        const _Float16* aH = aL + (size_t)16 * DDIM;
#pragma unroll
        for (int kt = 0; kt < 16; ++kt) {
            const int koff = kt * 32 + hi * 16;
            const v8h lo0 = *(const v8h*)(aL + koff);
            const v8h up0 = *(const v8h*)(aL + koff + 8);
            const v8h lo1 = *(const v8h*)(aH + koff);
            const v8h up1 = *(const v8h*)(aH + koff + 8);
            v16h a0, a1;
#pragma unroll
            for (int i = 0; i < 8; ++i) {
                a0[i] = lo0[i]; a0[i + 8] = up0[i];
                a1[i] = lo1[i]; a1[i + 8] = up1[i];
            }
            afragL[kt] = a0; afragH[kt] = a1;
        }
    }

    // Online softmax state per (lane, j) for both strips.
    float mA[8], sA[8], pA[8], mB[8], sB[8], pB[8];
#pragma unroll
    for (int j = 0; j < 8; ++j) {
        mA[j] = -1e30f; sA[j] = 0.0f; pA[j] = 0.0f;
        mB[j] = -1e30f; sB[j] = 0.0f; pB[j] = 0.0f;
    }

    // Async stage of one 32-column tile: wave w copies emb rows r0..r0+7
    // (1024B each = 64 x 16B chunks; lanes take chunks lane and lane+32).
    auto stage = [&](int ntbase, int buf) {
        const int r0 = wave * 8;
#pragma unroll
        for (int r = 0; r < 8; ++r) {
            const uint64_t g = (uint64_t)(uintptr_t)
                ((const char*)(emb + (size_t)(ntbase + r0 + r) * DDIM)) + (uint32_t)(lane * 16);
            const uint32_t l = (uint32_t)(uintptr_t)(&btile[buf][(r0 + r) * BPAD])
                               + (uint32_t)(lane * 16);
            async_load16(l, g);
            async_load16(l + 512, g + 512);
        }
    };

    stage(cbase, 0);
    wait_async0();
    __syncthreads();

    for (int nt = 0; nt < NSTAGES; ++nt) {
        const int cur = nt & 1;
        if (nt + 1 < NSTAGES) stage(cbase + (nt + 1) * TILE_N, cur ^ 1);  // overlap

#pragma unroll
        for (int sub = 0; sub < 2; ++sub) {
            // B 32x16 f16 operand, k order used here (same as A): lane n = column n, 16 contiguous halfs at
            // K [k0 + 16*hi .. +15] of the staged emb row.
            const _Float16* bptr = &btile[cur][(sub * 16 + l15) * BPAD + hi * 16];

            v8f accA = {}, accB = {};
            v16h b = *(const v16h*)(bptr);          // kt = 0
#pragma unroll
            for (int kt = 0; kt < 16; ++kt) {
                const v16h bn = (kt < 15) ? *(const v16h*)(bptr + (kt + 1) * 32) : b;
                accA = __builtin_amdgcn_wmma_f32_16x16x32_f16(
                    false, afragL[kt], false, b, (short)0, accA, false, false);
                accB = __builtin_amdgcn_wmma_f32_16x16x32_f16(
                    false, afragH[kt], false, b, (short)0, accB, false, false);
                b = bn;
            }

            // C layout: VGPR j = row M=j (lanes 0-15) / M=j+8 (lanes 16-31),
            // column N = lane&15.
            const int gj = cbase + nt * TILE_N + sub * 16 + l15;
#pragma unroll
            for (int j = 0; j < 8; ++j) {
                {   // strip L
                    const int gi  = mblk + wave * 32 + hi * 8 + j;
                    float logit   = accA[j] * INV_T;
                    const int tgt = (gi < NHALF) ? (gi + NHALF) : (gi - NHALF);
                    if (gj == tgt) pA[j] += logit;
                    if (gi == gj)  logit = -1e30f;
                    const float mn = fmaxf(mA[j], logit);
                    sA[j] = sA[j] * __expf(mA[j] - mn) + __expf(logit - mn);
                    mA[j] = mn;
                }
                {   // strip H
                    const int gi  = mblk + wave * 32 + 16 + hi * 8 + j;
                    float logit   = accB[j] * INV_T;
                    const int tgt = (gi < NHALF) ? (gi + NHALF) : (gi - NHALF);
                    if (gj == tgt) pB[j] += logit;
                    if (gi == gj)  logit = -1e30f;
                    const float mn = fmaxf(mB[j], logit);
                    sB[j] = sB[j] * __expf(mB[j] - mn) + __expf(logit - mn);
                    mB[j] = mn;
                }
            }
        }

        wait_async0();      // next tile fully landed in LDS (this wave)
        __syncthreads();    // ... and for all waves; prev buffer reusable
    }

    // Reduce (m, s, pos) across the 16 lanes sharing each row
    // (xor masks 1,2,4,8 stay within each 16-lane half of the wave32).
#pragma unroll
    for (int j = 0; j < 8; ++j) {
#pragma unroll
        for (int strip = 0; strip < 2; ++strip) {
            float m = strip ? mB[j] : mA[j];
            float s = strip ? sB[j] : sA[j];
            float p = strip ? pB[j] : pA[j];
#pragma unroll
            for (int off = 1; off < 16; off <<= 1) {
                const float m2 = __shfl_xor(m, off, 32);
                const float s2 = __shfl_xor(s, off, 32);
                const float p2 = __shfl_xor(p, off, 32);
                const float mn = fmaxf(m, m2);
                s = s * __expf(m - mn) + s2 * __expf(m2 - mn);
                m = mn;
                p += p2;
            }
            if (strip) { mB[j] = m; sB[j] = s; pB[j] = p; }
            else       { mA[j] = m; sA[j] = s; pA[j] = p; }
        }
    }

    // Lanes 0 and 16 own 8 rows per strip: write per-row partials.
    if (l15 == 0) {
        const size_t base = (size_t)chalf * NROWS;
#pragma unroll
        for (int j = 0; j < 8; ++j) {
            const int rL = mblk + wave * 32 + hi * 8 + j;
            const int rH = rL + 16;
            pm[base + rL] = mA[j]; ps[base + rL] = sA[j]; pp[base + rL] = pA[j];
            pm[base + rH] = mB[j]; ps[base + rH] = sB[j]; pp[base + rH] = pB[j];
        }
    }
}

// ---------------------------------------------------------------------------
// Kernel 4: merge the two column-half partials per row and reduce to loss.
// ---------------------------------------------------------------------------
__global__ void __launch_bounds__(256)
merge_loss(const float* __restrict__ pm, const float* __restrict__ ps,
           const float* __restrict__ pp, float* __restrict__ out) {
    const int r  = blockIdx.x * 256 + threadIdx.x;
    const float m0 = pm[r],         m1 = pm[NROWS + r];
    const float s0 = ps[r],         s1 = ps[NROWS + r];
    const float p  = pp[r] + pp[NROWS + r];
    const float mm = fmaxf(m0, m1);
    const float s  = s0 * __expf(m0 - mm) + s1 * __expf(m1 - mm);
    float loss = mm + logf(s) - p;
#pragma unroll
    for (int off = 16; off > 0; off >>= 1)
        loss += __shfl_xor(loss, off, 32);
    if ((threadIdx.x & 31) == 0)
        atomicAdd(out, loss * (1.0f / (float)NROWS));
}

// ---------------------------------------------------------------------------
extern "C" void kernel_launch(void* const* d_in, const int* in_sizes, int n_in,
                              void* d_out, int out_size, void* d_ws, size_t ws_size,
                              hipStream_t stream) {
    const float* z1 = (const float*)d_in[0];
    const float* z2 = (const float*)d_in[1];
    float* out      = (float*)d_out;
    _Float16* emb   = (_Float16*)d_ws;                 // 8 MB
    float* pm = (float*)((char*)d_ws + EMB_BYTES);     // [2][NROWS]
    float* ps = pm + COLSPLIT * NROWS;
    float* pp = ps + COLSPLIT * NROWS;

    l2norm_to_f16<<<NROWS, 128, 0, stream>>>(z1, z2, emb);
    dim3 grid(NROWS / 128, COLSPLIT);
    ntxent_fused<<<grid, 128, 0, stream>>>(emb, pm, ps, pp);
    zero_scalar<<<1, 1, 0, stream>>>(out);
    merge_loss<<<NROWS / 256, 256, 0, stream>>>(pm, ps, pp, out);
}
